// GNN_70961449664571
// MI455X (gfx1250) — matrix operand order repaired and checked by reading, not compiled, not run on hardware
//
#include <hip/hip_runtime.h>

typedef __attribute__((ext_vector_type(16))) __bf16 v16bf;
typedef __attribute__((ext_vector_type(2)))  __bf16 v2bf;
typedef __attribute__((ext_vector_type(8)))  float  v8f;

#define N_NODES 50000
#define N_PAD   50048   // ((N + 127)/128)*128 : row-padded workspace so GEMM A loads need no guard

__device__ __forceinline__ unsigned short f2bf_bits(float f) {
    union { float f; unsigned int u; } cv; cv.f = f;
    unsigned int u = cv.u;
    u += 0x7FFFu + ((u >> 16) & 1u);   // round-to-nearest-even
    return (unsigned short)(u >> 16);
}

// Pack two fp32 -> two bf16 in one dword; hardware cvt if the toolchain exposes it.
__device__ __forceinline__ unsigned int pack2_bf16(float f0, float f1) {
#if __has_builtin(__builtin_amdgcn_cvt_pk_bf16_f32)
    union { v2bf v; unsigned int u; } cv;
    cv.v = __builtin_amdgcn_cvt_pk_bf16_f32(f0, f1);
    return cv.u;
#else
    return (unsigned int)f2bf_bits(f0) | ((unsigned int)f2bf_bits(f1) << 16);
#endif
}

// ---------------- elementwise helpers ----------------

__global__ void copy_f32x4(const float* __restrict__ src, float* __restrict__ dst, int n4) {
    int i = blockIdx.x * blockDim.x + threadIdx.x;
    if (i < n4) ((float4*)dst)[i] = ((const float4*)src)[i];
}

__global__ void f32_to_bf16(const float* __restrict__ in, unsigned short* __restrict__ out, int n) {
    int i = blockIdx.x * blockDim.x + threadIdx.x;
    if (i < n) out[i] = f2bf_bits(in[i]);
}

// One wave per edge: lanes cover the feature dimension, coalesced atomic adds.
// Edge indices are wave-uniform: force them scalar so the index fetch is an s_load
// and the row base addresses live in SGPRs.
__global__ void scatter_add_edges(const float* __restrict__ x, const int* __restrict__ edges,
                                  float* __restrict__ h, int E, int d) {
    int wid  = (blockIdx.x * blockDim.x + threadIdx.x) >> 5;
    int lane = threadIdx.x & 31;
    int nw   = (gridDim.x * blockDim.x) >> 5;
    for (int e0 = wid; e0 < E; e0 += nw) {
        const int e = __builtin_amdgcn_readfirstlane(e0);
        const int s = __builtin_amdgcn_readfirstlane(edges[e]);      // source node
        const int t = __builtin_amdgcn_readfirstlane(edges[E + e]);  // destination node
        const float* xs = x + (size_t)s * d;
        float*       hd = h + (size_t)t * d;
        for (int j = lane; j < d; j += 32)
            atomicAdd(&hd[j], xs[j]);
    }
}

// ---------------- WMMA GEMM: out = relu(A[N_PAD,K] @ B[K,ncols] + bias) ----------------
// Block: 256 threads = 8 waves. Block tile: 128 rows x 64 cols.
// Each wave: one 16-row strip x 64 cols (4 WMMA accumulators sharing one A fragment).
// B panel staged in LDS TRANSPOSED (column-major, padded stride) so a lane's
// 16-bf16 B fragment is two 16-byte runs 32 bytes apart -> two ds_load_b128, no repacking.
// A is always a row-padded workspace buffer: loads are unconditional (no EXEC games);
// only the final store is guarded by m < N.

#define KST_PAD 8   // pad ushorts per column: keeps 16B alignment, breaks bank aliasing

template <bool A_IS_BF16, bool OUT_IS_BF16>
__global__ __launch_bounds__(256) void gemm_bias_relu(
    const void* __restrict__ Aptr, const unsigned short* __restrict__ B,
    const float* __restrict__ bias, void* __restrict__ outPtr,
    int N, int K, int ncols)
{
    // 64 cols x (256+8) ushorts = 33792 B max
    __shared__ unsigned int sBtu[(64 * (256 + KST_PAD)) / 2];
    unsigned short* sBt = (unsigned short*)sBtu;

    const int tid  = threadIdx.x;
    const int wave = tid >> 5;
    const int lane = tid & 31;
    const int col0 = blockIdx.y * 64;
    const int Kst  = K + KST_PAD;

    // ---- Stage B panel transposed: sBt[c*Kst + k] = B[k][col0 + c].
    // 2x2 register transpose: coalesced b32 global reads, b32 LDS stores.
    const int nblk = (K >> 1) * 32;            // (K/2) k-pairs x 32 col-pairs
    for (int idx = tid; idx < nblk; idx += 256) {
        const int k = (idx >> 5) << 1;         // even k
        const int c = (idx & 31) << 1;         // even col
        const unsigned int u0 = ((const unsigned int*)(B + (size_t)k       * ncols + col0))[idx & 31];
        const unsigned int u1 = ((const unsigned int*)(B + (size_t)(k + 1) * ncols + col0))[idx & 31];
        const unsigned int oc  = (u0 & 0xFFFFu) | (u1 << 16);          // (k,c),(k+1,c)
        const unsigned int oc1 = (u0 >> 16)     | (u1 & 0xFFFF0000u);  // (k,c+1),(k+1,c+1)
        sBtu[((c)     * Kst + k) >> 1] = oc;
        sBtu[((c + 1) * Kst + k) >> 1] = oc1;
    }
    __syncthreads();

    const int row  = blockIdx.x * 128 + wave * 16 + (lane & 15);  // always < N_PAD
    const int koff = (lane >> 4) * 8;                 // A lane K-offset (16x32 bf16 layout)
    const int kbB  = (lane >> 4) * 8;                 // B lane K-offset (same k order as A)
    const int cl   = lane & 15;

    // per-lane base into transposed panel: column cl (+16 per sub-tile), K-half kbB
    const unsigned short* pB = sBt + (size_t)cl * Kst + kbB;

    v8f acc[4];
    #pragma unroll
    for (int s = 0; s < 4; ++s)
        #pragma unroll
        for (int e = 0; e < 8; ++e) acc[s][e] = 0.0f;

    union Frag { uint4 v[2]; v16bf f; };

    for (int k0 = 0; k0 < K; k0 += 32) {
        // ---- A fragment: elems 0..7 = K(k0+koff..+7), elems 8..15 = K(k0+16+koff..+7)
        Frag a;
        if constexpr (A_IS_BF16) {
            const unsigned short* A16 = (const unsigned short*)Aptr;
            a.v[0] = *(const uint4*)(A16 + (size_t)row * K + k0 + koff);
            a.v[1] = *(const uint4*)(A16 + (size_t)row * K + k0 + 16 + koff);
        } else {
            const float* A32 = (const float*)Aptr;
            const float4* p0 = (const float4*)(A32 + (size_t)row * K + k0 + koff);
            const float4* p1 = (const float4*)(A32 + (size_t)row * K + k0 + 16 + koff);
            float4 q0 = p0[0], q1 = p0[1], q2 = p1[0], q3 = p1[1];
            float f01[16] = {q0.x, q0.y, q0.z, q0.w, q1.x, q1.y, q1.z, q1.w,
                             q2.x, q2.y, q2.z, q2.w, q3.x, q3.y, q3.z, q3.w};
            unsigned int* au = (unsigned int*)&a;
            #pragma unroll
            for (int j = 0; j < 8; ++j)
                au[j] = pack2_bf16(f01[2 * j], f01[2 * j + 1]);
        }

        // ---- 4 column sub-tiles: B fragment = two 16-byte LDS runs, k0+kbB.. and k0+16+kbB..
        #pragma unroll
        for (int s = 0; s < 4; ++s) {
            Frag b;
            const uint4* pb = (const uint4*)(pB + (size_t)(s * 16) * Kst + k0);
            b.v[0] = pb[0];
            b.v[1] = pb[2];
            acc[s] = __builtin_amdgcn_wmma_f32_16x16x32_bf16(
                false, a.f, false, b.f, (short)0, acc[s], false, false);
        }
    }

    // ---- epilogue: bias + relu, C/D layout: VGPR r -> row r + (lane>=16)*8
    const int rbase = (lane >> 4) * 8;
    const int mrow0 = blockIdx.x * 128 + wave * 16 + rbase;
    #pragma unroll
    for (int s = 0; s < 4; ++s) {
        const int n  = col0 + s * 16 + cl;
        const float bv = bias[n];
        #pragma unroll
        for (int r = 0; r < 8; ++r) {
            const int m = mrow0 + r;
            if (m < N) {
                float v = fmaxf(acc[s][r] + bv, 0.0f);
                if constexpr (OUT_IS_BF16)
                    ((unsigned short*)outPtr)[(size_t)m * ncols + n] = f2bf_bits(v);
                else
                    ((float*)outPtr)[(size_t)m * ncols + n] = v;
            }
        }
    }
}

// ---------------- host launcher ----------------

extern "C" void kernel_launch(void* const* d_in, const int* in_sizes, int n_in,
                              void* d_out, int out_size, void* d_ws, size_t ws_size,
                              hipStream_t stream) {
    const int N  = N_NODES;
    const int NP = N_PAD;
    const float* x     = (const float*)d_in[0];
    const int*   edges = (const int*)d_in[1];
    const float* W[8]  = {(const float*)d_in[2],  (const float*)d_in[4],
                          (const float*)d_in[6],  (const float*)d_in[8],
                          (const float*)d_in[10], (const float*)d_in[12],
                          (const float*)d_in[14], (const float*)d_in[16]};
    const float* bias[8] = {(const float*)d_in[3],  (const float*)d_in[5],
                            (const float*)d_in[7],  (const float*)d_in[9],
                            (const float*)d_in[11], (const float*)d_in[13],
                            (const float*)d_in[15], (const float*)d_in[17]};
    const int E = in_sizes[1] / 2;

    // workspace layout (row-padded to NP so GEMM A loads are unguarded)
    char* ws = (char*)d_ws;
    float*          h  = (float*)ws;                                    // NP*256 f32
    unsigned short* mb = (unsigned short*)(ws + (size_t)NP * 256 * 4);  // NP*256 bf16
    float*          yA = (float*)(ws + (size_t)NP * 256 * 4 + (size_t)NP * 256 * 2);
    float*          yB = (float*)((char*)yA + (size_t)NP * 256 * 4);    // NP*128 f32
    unsigned short* wb = (unsigned short*)((char*)yB + (size_t)NP * 128 * 4);

    const int wcnt[8] = {128 * 256, 256 * 256, 256 * 128, 128 * 128,
                         128 * 128, 128 * 128, 128 * 128, 128 * 128};
    unsigned short* wbp[8];
    {
        size_t off = 0;
        for (int i = 0; i < 8; ++i) {
            wbp[i] = wb + off;
            off += (size_t)wcnt[i];
        }
    }
    for (int i = 0; i < 8; ++i)
        f32_to_bf16<<<(wcnt[i] + 255) / 256, 256, 0, stream>>>(W[i], wbp[i], wcnt[i]);

    const int gm = NP / 128;                   // 391
    const dim3 blk(256);
    const int sgrid = (E + 7) / 8;             // one wave per edge

    // ---- Layer 1: d_in 128 -> 256 -> 256
    copy_f32x4<<<(N * 128 / 4 + 255) / 256, blk, 0, stream>>>(x, h, N * 128 / 4);
    scatter_add_edges<<<sgrid, blk, 0, stream>>>(x, edges, h, E, 128);
    gemm_bias_relu<false, true ><<<dim3(gm, 4), blk, 0, stream>>>(h,  wbp[0], bias[0], mb, N, 128, 256);
    gemm_bias_relu<true,  false><<<dim3(gm, 4), blk, 0, stream>>>(mb, wbp[1], bias[1], yA, N, 256, 256);

    // ---- Layer 2: 256 -> 128 -> 128
    copy_f32x4<<<(N * 256 / 4 + 255) / 256, blk, 0, stream>>>(yA, h, N * 256 / 4);
    scatter_add_edges<<<sgrid, blk, 0, stream>>>(yA, edges, h, E, 256);
    gemm_bias_relu<false, true ><<<dim3(gm, 2), blk, 0, stream>>>(h,  wbp[2], bias[2], mb, N, 256, 128);
    gemm_bias_relu<true,  false><<<dim3(gm, 2), blk, 0, stream>>>(mb, wbp[3], bias[3], yB, N, 128, 128);

    // ---- Layer 3: 128 -> 128 -> 128
    copy_f32x4<<<(N * 128 / 4 + 255) / 256, blk, 0, stream>>>(yB, h, N * 128 / 4);
    scatter_add_edges<<<sgrid, blk, 0, stream>>>(yB, edges, h, E, 128);
    gemm_bias_relu<false, true ><<<dim3(gm, 2), blk, 0, stream>>>(h,  wbp[4], bias[4], mb, N, 128, 128);
    gemm_bias_relu<true,  false><<<dim3(gm, 2), blk, 0, stream>>>(mb, wbp[5], bias[5], yA, N, 128, 128);

    // ---- Layer 4: 128 -> 128 -> 128, final output f32
    copy_f32x4<<<(N * 128 / 4 + 255) / 256, blk, 0, stream>>>(yA, h, N * 128 / 4);
    scatter_add_edges<<<sgrid, blk, 0, stream>>>(yA, edges, h, E, 128);
    gemm_bias_relu<false, true ><<<dim3(gm, 2), blk, 0, stream>>>(h,  wbp[6], bias[6], mb, N, 128, 128);
    gemm_bias_relu<true,  false><<<dim3(gm, 2), blk, 0, stream>>>(mb, wbp[7], bias[7], (float*)d_out, N, 128, 128);
}
